// GCN_8512625180874
// MI455X (gfx1250) — matrix operand order repaired and checked by reading, not compiled, not run on hardware
//
#include <hip/hip_runtime.h>
#include <hip/hip_bf16.h>

// ---------------------------------------------------------------------------
// GCN forward for MI455X (gfx1250, wave32).
// GEMMs: bf16 WMMA (v_wmma_f32_16x16x32_bf16, fp32 accumulate), 128x64 block
// tile, one 16-row stripe x four 16x16 tiles per wave, compile-time K/N so
// all tail guards fold. M is always a multiple of 16 here (N_NODES=50000),
// so C stores need only a per-tile column predicate.
// Edge aggregation (the real bottleneck, ~3 GB moved) uses float4 gathers +
// f32 atomics; the 40MB activation set lives in the 192MB L2.
// ---------------------------------------------------------------------------

typedef __attribute__((ext_vector_type(16))) __bf16 v16bf;
typedef __attribute__((ext_vector_type(8)))  __bf16 v8bf;
typedef __attribute__((ext_vector_type(2)))  __bf16 v2bf;
typedef __attribute__((ext_vector_type(8)))  float  v8f;

#define EMB 200
#define NFEAT 4
#define HID 128

__device__ __forceinline__ v2bf pack2(float a, float b) {
  v2bf r; r[0] = (__bf16)a; r[1] = (__bf16)b; return r;   // v_cvt_pk_bf16_f32
}

// C[M,N] = A[M,K] @ B[K,N]; fp32 in/out, bf16 WMMA inside. M % 16 == 0.
// Block = 256 threads = 8 wave32s. Block tile 128(M) x 64(N):
// wave w owns rows [w*16, w*16+16) and all four 16-wide N tiles.
template <int K, int N>
__global__ __launch_bounds__(256)
void gemm_wmma(const float* __restrict__ A, const float* __restrict__ Bm,
               float* __restrict__ C, int M) {
  constexpr int KT = 32;
  __shared__ __bf16 lsA[128 * KT];   // [row][k]          8 KB
  __shared__ __bf16 lsBT[64 * KT];   // transposed [n][k] 4 KB

  const int tid  = threadIdx.x;
  const int wave = tid >> 5;
  const int lane = tid & 31;
  const int hlf  = lane >> 4;
  const int l16  = lane & 15;

  const int mBlk  = blockIdx.x * 128;
  const int n0    = blockIdx.y * 64;
  const bool fullM = (mBlk + 128 <= M);
  const bool fullN = (n0 + 64 <= N);

  v8f acc[4] = {{}, {}, {}, {}};

  for (int k0 = 0; k0 < K; k0 += KT) {
    const int krem = K - k0;

    // ---- stage A panel 128x32 as packed bf16 pairs ----
    if (krem >= KT) {
#pragma unroll
      for (int t = 0; t < 8; ++t) {
        int p = tid + t * 256;                 // 2048 packs
        int r = p >> 4, kp = p & 15;
        int gr = mBlk + r;
        if (!fullM && gr >= M) gr = M - 1;     // clamp; never stored
        float2 v = *(const float2*)&A[(long)gr * K + k0 + 2 * kp];
        *(v2bf*)&lsA[r * KT + 2 * kp] = pack2(v.x, v.y);
      }
    } else {
#pragma unroll
      for (int t = 0; t < 8; ++t) {
        int p = tid + t * 256;
        int r = p >> 4, kp = p & 15, k = 2 * kp;
        int gr = mBlk + r;
        if (!fullM && gr >= M) gr = M - 1;
        const float* ap = &A[(long)gr * K + k0 + k];
        float f0 = (k < krem) ? ap[0] : 0.0f;
        float f1 = (k + 1 < krem) ? ap[1] : 0.0f;
        *(v2bf*)&lsA[r * KT + 2 * kp] = pack2(f0, f1);
      }
    }

    // ---- stage B panel 32x64, transposed to [n][k] ----
#pragma unroll
    for (int t = 0; t < 4; ++t) {
      int p = tid + t * 256;                   // 1024 packs
      int n = p >> 4, kp = p & 15, k = 2 * kp;
      int gn = n0 + n;
      float f0 = 0.0f, f1 = 0.0f;
      if (fullN || gn < N) {
        if (k < krem)     f0 = Bm[(long)(k0 + k) * N + gn];
        if (k + 1 < krem) f1 = Bm[(long)(k0 + k + 1) * N + gn];
      }
      *(v2bf*)&lsBT[n * KT + 2 * kp] = pack2(f0, f1);
    }

    // prefetch next A panel (global_prefetch_b8)
    if (k0 + KT < K) {
      int pr = mBlk + (tid & 127);
      if (pr >= M) pr = M - 1;
      __builtin_prefetch(&A[(long)pr * K + k0 + KT], 0, 1);
    }
    __syncthreads();

    // ---- fragments (ISA 7.12.2 layouts; contiguous b128 LDS loads) ----
    // A 16x32: lane m=l16 (both halves); elems 0..7 = k(8h..8h+7),
    //          elems 8..15 = k(16+8h..23+8h)  -> two 16B loads
    const __bf16* aRow = &lsA[(wave * 16 + l16) * KT];
    v8bf alo = *(const v8bf*)(aRow + 8 * hlf);
    v8bf ahi = *(const v8bf*)(aRow + 16 + 8 * hlf);
    v16bf af = __builtin_shufflevector(alo, ahi, 0, 1, 2, 3, 4, 5, 6, 7,
                                       8, 9, 10, 11, 12, 13, 14, 15);
#pragma unroll
    for (int t = 0; t < 4; ++t) {
      // B 32x16: lane n=l16; elems 0..7 = k(8h..8h+7), elems 8..15 = k(16+8h..23+8h) -> two 16B loads
      const __bf16* bRow = &lsBT[(t * 16 + l16) * KT];
      v8bf blo = *(const v8bf*)(bRow + 8 * hlf);
      v8bf bhi = *(const v8bf*)(bRow + 16 + 8 * hlf);
      v16bf bfr = __builtin_shufflevector(blo, bhi, 0, 1, 2, 3, 4, 5, 6, 7,
                                          8, 9, 10, 11, 12, 13, 14, 15);
      acc[t] = __builtin_amdgcn_wmma_f32_16x16x32_bf16(
          false, af, false, bfr, (short)0, acc[t], false, false);
    }
    __syncthreads();
  }

  // ---- store: D VGPR r holds M = r + 8*half, N = l16 ----
  // M % 16 == 0 -> all 8 rows valid whenever m0 < M; only columns predicated.
  const int m0 = mBlk + wave * 16;
  if (m0 >= M) return;
  float* cBase = &C[(long)(m0 + 8 * hlf) * N + n0 + l16];
#pragma unroll
  for (int t = 0; t < 4; ++t) {
    if (n0 + t * 16 + l16 < N) {               // per-tile lane predicate
#pragma unroll
      for (int r = 0; r < 8; ++r) {
        cBase[(long)r * N + t * 16] = acc[t][r];
      }
    }
  }
}

// ---------------------------------------------------------------------------

__global__ void fill_f32(float* __restrict__ p, float v, int n) {
  int i = blockIdx.x * blockDim.x + threadIdx.x;
  if (i < n) p[i] = v;
}

__global__ void deg_count(const int* __restrict__ ei, float* __restrict__ deg, int E) {
  int e = blockIdx.x * blockDim.x + threadIdx.x;
  if (e < E) atomicAdd(&deg[ei[E + e]], 1.0f);
}

__global__ void rsqrt_k(const float* __restrict__ deg, float* __restrict__ dinv, int n) {
  int i = blockIdx.x * blockDim.x + threadIdx.x;
  if (i < n) dinv[i] = rsqrtf(deg[i]);
}

__global__ void edge_norm(const int* __restrict__ ei, const float* __restrict__ dinv,
                          float* __restrict__ nrm, int E) {
  int e = blockIdx.x * blockDim.x + threadIdx.x;
  if (e < E) nrm[e] = dinv[ei[e]] * dinv[ei[E + e]];
}

// out[n,f] = bias[f] + hw[n,f] * dinv[n]^2   (self-loop + bias)
__global__ void self_init(const float* __restrict__ hw, const float* __restrict__ dinv,
                          const float* __restrict__ bias, float* __restrict__ out, int Nn) {
  int i = blockIdx.x * blockDim.x + threadIdx.x;
  if (i < Nn * EMB) {
    int n = i / EMB, f = i - n * EMB;
    float d = dinv[n];
    out[i] = bias[f] + hw[i] * d * d;
  }
}

// out[dst,:] += hw[src,:] * norm[e]
__global__ void edge_scatter(const int* __restrict__ ei, const float* __restrict__ nrm,
                             const float* __restrict__ hw, float* __restrict__ out, int E) {
  constexpr int C4 = EMB / 4;
  int i = blockIdx.x * blockDim.x + threadIdx.x;
  if (i >= E * C4) return;
  int e = i / C4;
  int c = (i - e * C4) << 2;
  int s = ei[e], d = ei[E + e];
  float w = nrm[e];
  float4 v = *(const float4*)&hw[(long)s * EMB + c];
  float* o = &out[(long)d * EMB + c];
  atomicAdd(o + 0, v.x * w);
  atomicAdd(o + 1, v.y * w);
  atomicAdd(o + 2, v.z * w);
  atomicAdd(o + 3, v.w * w);
}

__global__ void leaky_k(float* __restrict__ p, int n) {
  int i = blockIdx.x * blockDim.x + threadIdx.x;
  if (i < n) {
    float v = p[i];
    p[i] = v > 0.0f ? v : 0.01f * v;
  }
}

__global__ void pool_sum(const float* __restrict__ h, const int* __restrict__ batch,
                         float* __restrict__ sums, int Nn) {
  constexpr int C4 = EMB / 4;
  int i = blockIdx.x * blockDim.x + threadIdx.x;
  if (i >= Nn * C4) return;
  int n = i / C4;
  int c = (i - n * C4) << 2;
  int g = batch[n];
  float4 v = *(const float4*)&h[(long)n * EMB + c];
  float* o = &sums[(long)g * EMB + c];
  atomicAdd(o + 0, v.x);
  atomicAdd(o + 1, v.y);
  atomicAdd(o + 2, v.z);
  atomicAdd(o + 3, v.w);
}

__global__ void pool_cnt(const int* __restrict__ batch, float* __restrict__ cnt, int Nn) {
  int n = blockIdx.x * blockDim.x + threadIdx.x;
  if (n < Nn) atomicAdd(&cnt[batch[n]], 1.0f);
}

__global__ void build_g(const float* __restrict__ sums, const float* __restrict__ cnt,
                        const float* __restrict__ xs, float* __restrict__ g, int G) {
  constexpr int FC = EMB + NFEAT;
  int i = blockIdx.x * blockDim.x + threadIdx.x;
  if (i >= G * FC) return;
  int gi = i / FC, f = i - gi * FC;
  g[i] = (f < EMB) ? sums[gi * EMB + f] / fmaxf(cnt[gi], 1.0f)
                   : xs[gi * NFEAT + (f - EMB)];
}

__global__ __launch_bounds__(HID)
void mlp_head(const float* __restrict__ g, const float* __restrict__ Wl1,
              const float* __restrict__ bl1, const float* __restrict__ Wl2,
              const float* __restrict__ bl2, float* __restrict__ out) {
  constexpr int FC = EMB + NFEAT;
  __shared__ float red[HID];
  int gi = blockIdx.x, t = threadIdx.x;
  float acc = bl1[t];
  const float* gr = &g[gi * FC];
  for (int k = 0; k < FC; ++k) acc += gr[k] * Wl1[k * HID + t];
  acc = acc > 0.0f ? acc : 0.01f * acc;
  red[t] = acc * Wl2[t];
  __syncthreads();
  for (int s = HID / 2; s > 0; s >>= 1) {
    if (t < s) red[t] += red[t + s];
    __syncthreads();
  }
  if (t == 0) out[gi] = red[0] + bl2[0];
}

// ---------------------------------------------------------------------------

extern "C" void kernel_launch(void* const* d_in, const int* in_sizes, int n_in,
                              void* d_out, int out_size, void* d_ws, size_t ws_size,
                              hipStream_t stream) {
  const float* x     = (const float*)d_in[0];
  const int*   ei    = (const int*)  d_in[1];
  const float* xs    = (const float*)d_in[2];
  const int*   batch = (const int*)  d_in[3];
  const float* W0 = (const float*)d_in[4];  const float* b0 = (const float*)d_in[5];
  const float* W1 = (const float*)d_in[6];  const float* b1 = (const float*)d_in[7];
  const float* W2 = (const float*)d_in[8];  const float* b2 = (const float*)d_in[9];
  const float* W3 = (const float*)d_in[10]; const float* b3 = (const float*)d_in[11];
  const float* Wl1 = (const float*)d_in[12]; const float* bl1 = (const float*)d_in[13];
  const float* Wl2 = (const float*)d_in[14]; const float* bl2 = (const float*)d_in[15];
  float* out = (float*)d_out;

  const int Nn = in_sizes[0] / NFEAT;   // 50000 (multiple of 16)
  const int E  = in_sizes[1] / 2;       // 400000
  const int G  = in_sizes[2] / NFEAT;   // 500

  char* wsp = (char*)d_ws;
  auto carve = [&](size_t bytes) {
    char* p = wsp;
    wsp += (bytes + 255) & ~(size_t)255;
    return p;
  };
  float* deg  = (float*)carve((size_t)Nn * 4);
  float* dinv = (float*)carve((size_t)Nn * 4);
  float* nrm  = (float*)carve((size_t)E * 4);
  float* bufA = (float*)carve((size_t)Nn * EMB * 4);
  float* bufB = (float*)carve((size_t)Nn * EMB * 4);
  float* hw   = (float*)carve((size_t)Nn * EMB * 4);
  float* sums = (float*)carve((size_t)G * EMB * 4);
  float* cnt  = (float*)carve((size_t)G * 4);
  float* gbuf = (float*)carve((size_t)G * (EMB + NFEAT) * 4);

  const int TB = 256;
  auto blocks = [&](long n) { return (unsigned)((n + TB - 1) / TB); };

  // degrees (self-loop contributes 1) and symmetric norms
  fill_f32<<<blocks(Nn), TB, 0, stream>>>(deg, 1.0f, Nn);
  deg_count<<<blocks(E), TB, 0, stream>>>(ei, deg, E);
  rsqrt_k<<<blocks(Nn), TB, 0, stream>>>(deg, dinv, Nn);
  edge_norm<<<blocks(E), TB, 0, stream>>>(ei, dinv, nrm, E);

  const dim3 ggrid((unsigned)((Nn + 127) / 128), (EMB + 63) / 64);
  const int tot = Nn * EMB;

  auto post_gemm = [&](const float* b, float* dst) {
    self_init<<<blocks(tot), TB, 0, stream>>>(hw, dinv, b, dst, Nn);
    edge_scatter<<<blocks((long)E * (EMB / 4)), TB, 0, stream>>>(ei, nrm, hw, dst, E);
    leaky_k<<<blocks(tot), TB, 0, stream>>>(dst, tot);
  };

  gemm_wmma<NFEAT, EMB><<<ggrid, 256, 0, stream>>>(x, W0, hw, Nn);
  post_gemm(b0, bufA);
  gemm_wmma<EMB, EMB><<<ggrid, 256, 0, stream>>>(bufA, W1, hw, Nn);
  post_gemm(b1, bufB);
  gemm_wmma<EMB, EMB><<<ggrid, 256, 0, stream>>>(bufB, W2, hw, Nn);
  post_gemm(b2, bufA);
  gemm_wmma<EMB, EMB><<<ggrid, 256, 0, stream>>>(bufA, W3, hw, Nn);
  post_gemm(b3, bufB);                       // final node features in bufB

  // global mean pool + concat + MLP head
  fill_f32<<<blocks(G * EMB), TB, 0, stream>>>(sums, 0.0f, G * EMB);
  fill_f32<<<blocks(G), TB, 0, stream>>>(cnt, 0.0f, G);
  pool_sum<<<blocks((long)Nn * (EMB / 4)), TB, 0, stream>>>(bufB, batch, sums, Nn);
  pool_cnt<<<blocks(Nn), TB, 0, stream>>>(batch, cnt, Nn);
  build_g<<<blocks(G * (EMB + NFEAT)), TB, 0, stream>>>(sums, cnt, xs, gbuf, G);
  mlp_head<<<G, HID, 0, stream>>>(gbuf, Wl1, bl1, Wl2, bl2, out);

  (void)n_in; (void)out_size; (void)ws_size;
}
